// ProbSparseSelfAttention_23149873725440
// MI455X (gfx1250) — matrix operand order repaired and checked by reading, not compiled, not run on hardware
//
#include <hip/hip_runtime.h>
#include <hip/hip_bf16.h>
#include <math.h>

typedef __attribute__((ext_vector_type(16))) __bf16 v16bf;
typedef __attribute__((ext_vector_type(8)))  float  v8f;

#define B_   4
#define L_   4096
#define D_   1024
#define H_   16
#define DH_  64
#define U_   45
#define BH_  (B_ * H_)

// ---------------- WMMA GEMM: C[M,N] = A[M,K] @ W[N,K]^T + bias[N] -------------
// M=16384, N=K=1024.  Block = 256 threads = 8 waves.
// Block tile 128(M) x 64(N); each wave owns a 16(M) x 64(N) strip:
// one A fragment reused across 4 v_wmma issues per K-chunk (KC=32).
// Tile staging uses CDNA5 async global->LDS copies (ASYNCcnt).
#define GM 16384
#define GN 1024
#define GK 1024
#define BM 128
#define BN 64
#define KC 32
#define ASTR 33   // padded LDS row stride (floats) for A tile
#define BSTR 33   // padded LDS row stride (floats) for W tile

__global__ __launch_bounds__(256)
void gemm_bf16_wmma(const float* __restrict__ A, const float* __restrict__ W,
                    const float* __restrict__ bias, float* __restrict__ C) {
  __shared__ float As[BM * ASTR];   // 128x32 fp32 tile (padded)  ~16.9 KB
  __shared__ float Ws[BN * BSTR];   //  64x32 fp32 tile (padded)   ~8.5 KB

  const int tid  = threadIdx.x;
  const int wave = tid >> 5;         // 0..7 -> M strip
  const int lane = tid & 31;
  const int half = lane >> 4;        // 16-lane half of the wave
  const int r    = lane & 15;
  const int m0   = blockIdx.y * BM;
  const int n0   = blockIdx.x * BN;

  v8f acc0 = {}, acc1 = {}, acc2 = {}, acc3 = {};

  for (int kc = 0; kc < GK; kc += KC) {
    // ---- async stage A tile (128x32 floats) : global -> LDS, no VGPR data ----
    const float* Abase = A + (size_t)m0 * GK + kc;
    #pragma unroll
    for (int i = 0; i < (BM * KC) / 256; ++i) {   // 16 issues/thread
      int idx = tid + i * 256;
      int rr = idx >> 5, cc = idx & 31;
      unsigned lds  = (unsigned)(size_t)&As[rr * ASTR + cc];
      unsigned goff = (unsigned)((rr * GK + cc) * 4);
      asm volatile("global_load_async_to_lds_b32 %0, %1, %2"
                   :: "v"(lds), "v"(goff), "s"(Abase) : "memory");
    }
    // ---- async stage W tile (64x32 floats) ----
    const float* Wbase = W + (size_t)n0 * GK + kc;
    #pragma unroll
    for (int i = 0; i < (BN * KC) / 256; ++i) {   // 8 issues/thread
      int idx = tid + i * 256;
      int rr = idx >> 5, cc = idx & 31;
      unsigned lds  = (unsigned)(size_t)&Ws[rr * BSTR + cc];
      unsigned goff = (unsigned)((rr * GK + cc) * 4);
      asm volatile("global_load_async_to_lds_b32 %0, %1, %2"
                   :: "v"(lds), "v"(goff), "s"(Wbase) : "memory");
    }
    asm volatile("s_wait_asynccnt 0x0" ::: "memory");
    __syncthreads();

    // ---- A fragment (CDNA5 16-bit A layout): lane r = row, j -> K index ----
    v16bf af;
    #pragma unroll
    for (int j = 0; j < 16; ++j) {
      int ka = ((j < 8) ? j : j + 8) + 8 * half;
      af[j] = (__bf16)As[(wave * 16 + r) * ASTR + ka];
    }
    // ---- sweep 4 N-tiles, reusing the A fragment ----
    #pragma unroll
    for (int nt = 0; nt < 4; ++nt) {
      v16bf bfv;
      #pragma unroll
      for (int j = 0; j < 16; ++j)
        bfv[j] = (__bf16)Ws[(nt * 16 + r) * BSTR + ((j < 8) ? j : j + 8) + 8 * half];
      v8f* accp = (nt == 0) ? &acc0 : (nt == 1) ? &acc1 : (nt == 2) ? &acc2 : &acc3;
      *accp = __builtin_amdgcn_wmma_f32_16x16x32_bf16(
          false, af, false, bfv, (short)0, *accp, false, false);
    }
    __syncthreads();
  }

  // C/D layout: VGPR v -> row M = v + 8*half, col N = r (per 16x16 tile)
  #pragma unroll
  for (int nt = 0; nt < 4; ++nt) {
    const v8f acc = (nt == 0) ? acc0 : (nt == 1) ? acc1 : (nt == 2) ? acc2 : acc3;
    const int n = n0 + nt * 16 + r;
    const float bv = bias[n];
    #pragma unroll
    for (int v = 0; v < 8; ++v) {
      int m = m0 + wave * 16 + v + 8 * half;
      C[(size_t)m * GN + n] = acc[v] + bv;
    }
  }
}

// --------- Sparsity measure: M[bh,q] = max_s(Q.Ksamp) - mean_s(Q.Ksamp) ------
__global__ __launch_bounds__(256)
void sparsity_measure(const float* __restrict__ Q, const float* __restrict__ K,
                      const long long* __restrict__ sidx, float* __restrict__ M) {
  const int bh = blockIdx.y;
  const int b  = bh / H_;
  const int h  = bh % H_;
  const int t  = threadIdx.x;

  __shared__ float Ks[U_][DH_];
  for (int i = t; i < U_ * DH_; i += 256) {
    int s = i / DH_, d = i % DH_;
    long long l = sidx[s];
    Ks[s][d] = K[((size_t)b * L_ + (size_t)l) * D_ + h * DH_ + d];
  }
  __syncthreads();

  const int q = blockIdx.x * 256 + t;
  const float* qrow = Q + ((size_t)b * L_ + q) * D_ + h * DH_;
  float qr[DH_];
  #pragma unroll
  for (int d = 0; d < DH_; ++d) qr[d] = qrow[d];

  float mx = -INFINITY, sm = 0.f;
  for (int s = 0; s < U_; ++s) {
    float acc = 0.f;
    #pragma unroll
    for (int d = 0; d < DH_; ++d) acc += qr[d] * Ks[s][d];
    mx = fmaxf(mx, acc);
    sm += acc;
  }
  M[(size_t)bh * L_ + q] = mx - sm * (1.0f / U_);
}

// ----------------- top-45 per (b,h): iterative argmax in LDS -----------------
__global__ __launch_bounds__(256)
void topk_u(const float* __restrict__ M, int* __restrict__ Mtop) {
  const int bh = blockIdx.x;
  const int t  = threadIdx.x;
  __shared__ float ms[L_];
  __shared__ float rv[256];
  __shared__ int   ri[256];
  for (int i = t; i < L_; i += 256) ms[i] = M[(size_t)bh * L_ + i];
  __syncthreads();
  for (int it = 0; it < U_; ++it) {
    float bv = -INFINITY; int bi = L_;
    for (int i = t; i < L_; i += 256) {
      float v = ms[i];
      if (v > bv || (v == bv && i < bi)) { bv = v; bi = i; }
    }
    rv[t] = bv; ri[t] = bi;
    __syncthreads();
    for (int s = 128; s > 0; s >>= 1) {
      if (t < s) {
        float ov = rv[t + s]; int oi = ri[t + s];
        if (ov > rv[t] || (ov == rv[t] && oi < ri[t])) { rv[t] = ov; ri[t] = oi; }
      }
      __syncthreads();
    }
    if (t == 0) { Mtop[bh * U_ + it] = ri[0]; ms[ri[0]] = -INFINITY; }
    __syncthreads();
  }
}

// -------- scores (raw, pre-softmax) written into the attn output region -----
__global__ __launch_bounds__(256)
void scores_kernel(const float* __restrict__ Q, const float* __restrict__ K,
                   const int* __restrict__ Mtop, float* __restrict__ attn) {
  const int bh = blockIdx.y;
  const int b = bh / H_, h = bh % H_;
  const int k0 = blockIdx.x * 256;
  const int t  = threadIdx.x;

  __shared__ float Qs[U_][DH_];
  for (int i = t; i < U_ * DH_; i += 256) {
    int j = i / DH_, d = i % DH_;
    int l = Mtop[bh * U_ + j];
    Qs[j][d] = Q[((size_t)b * L_ + l) * D_ + h * DH_ + d];
  }
  __syncthreads();

  float kr[DH_];
  const float* krow = K + ((size_t)b * L_ + k0 + t) * D_ + h * DH_;
  #pragma unroll
  for (int d = 0; d < DH_; ++d) kr[d] = krow[d];

  const float scale = 0.125f;  // 1/sqrt(64)
  for (int j = 0; j < U_; ++j) {
    float acc = 0.f;
    #pragma unroll
    for (int d = 0; d < DH_; ++d) acc += Qs[j][d] * kr[d];
    attn[((size_t)bh * U_ + j) * L_ + k0 + t] = acc * scale;
  }
}

// -------------------------- softmax in place over k --------------------------
__global__ __launch_bounds__(256)
void softmax_kernel(float* __restrict__ attn) {
  float* p = attn + (size_t)blockIdx.x * L_;
  const int t = threadIdx.x;
  __shared__ float buf[L_];
  __shared__ float red[256];
  float mx = -INFINITY;
  for (int i = t; i < L_; i += 256) { float v = p[i]; buf[i] = v; mx = fmaxf(mx, v); }
  red[t] = mx; __syncthreads();
  for (int s = 128; s > 0; s >>= 1) { if (t < s) red[t] = fmaxf(red[t], red[t + s]); __syncthreads(); }
  mx = red[0]; __syncthreads();
  float sm = 0.f;
  for (int i = t; i < L_; i += 256) { float e = expf(buf[i] - mx); buf[i] = e; sm += e; }
  red[t] = sm; __syncthreads();
  for (int s = 128; s > 0; s >>= 1) { if (t < s) red[t] += red[t + s]; __syncthreads(); }
  const float inv = 1.0f / red[0];
  for (int i = t; i < L_; i += 256) p[i] = buf[i] * inv;
}

// ------------------- context[bh,j,:] = attn[bh,j,:] @ V ----------------------
__global__ __launch_bounds__(256)
void context_kernel(const float* __restrict__ attn, const float* __restrict__ V,
                    float* __restrict__ ctx) {
  const int bh = blockIdx.y;
  const int j  = blockIdx.x;
  const int b = bh / H_, h = bh % H_;
  const int t = threadIdx.x;
  const int d = t & 63;
  const int part = t >> 6;  // split-K over 4 partitions
  const float* arow = attn + ((size_t)bh * U_ + j) * L_;
  float acc = 0.f;
  for (int k = part; k < L_; k += 4)
    acc += arow[k] * V[((size_t)b * L_ + k) * D_ + h * DH_ + d];
  __shared__ float red[256];
  red[t] = acc; __syncthreads();
  if (part == 0)
    ctx[((size_t)bh * U_ + j) * DH_ + d] = red[d] + red[64 + d] + red[128 + d] + red[192 + d];
}

// ----------------------------- V mean over L ---------------------------------
__global__ __launch_bounds__(256)
void vmean_kernel(const float* __restrict__ V, float* __restrict__ vmean) {
  const int bh = blockIdx.x;
  const int b = bh / H_, h = bh % H_;
  const int t = threadIdx.x;
  const int d = t & 63;
  const int part = t >> 6;
  float acc = 0.f;
  for (int l = part; l < L_; l += 4)
    acc += V[((size_t)b * L_ + l) * D_ + h * DH_ + d];
  __shared__ float red[256];
  red[t] = acc; __syncthreads();
  if (part == 0)
    vmean[bh * DH_ + d] = (red[d] + red[64 + d] + red[128 + d] + red[192 + d]) * (1.0f / L_);
}

// --------- default-fill with V_mean, then scatter selected contexts ----------
__global__ __launch_bounds__(256)
void fill_mean(const float* __restrict__ vmean, float* __restrict__ P) {
  size_t i = (size_t)blockIdx.x * 256 + threadIdx.x;   // over B*L*D
  int c = (int)(i % D_);
  size_t bl = i / D_;
  int b = (int)(bl / L_);
  int h = c / DH_, d = c % DH_;
  P[i] = vmean[(b * H_ + h) * DH_ + d];
}

__global__ __launch_bounds__(256)
void scatter_ctx(const float* __restrict__ ctx, const int* __restrict__ Mtop,
                 float* __restrict__ P) {
  int i = blockIdx.x * 256 + threadIdx.x;   // over BH*U*DH
  if (i >= BH_ * U_ * DH_) return;
  int d  = i % DH_;
  int j  = (i / DH_) % U_;
  int bh = i / (DH_ * U_);
  int b = bh / H_, h = bh % H_;
  int l = Mtop[bh * U_ + j];
  P[((size_t)b * L_ + l) * D_ + h * DH_ + d] = ctx[i];
}

// -----------------------------------------------------------------------------
extern "C" void kernel_launch(void* const* d_in, const int* in_sizes, int n_in,
                              void* d_out, int out_size, void* d_ws, size_t ws_size,
                              hipStream_t stream) {
  const float* queries = (const float*)d_in[0];
  const float* keys    = (const float*)d_in[1];
  const float* values  = (const float*)d_in[2];
  const float* Wq = (const float*)d_in[3];  const float* bq = (const float*)d_in[4];
  const float* Wk = (const float*)d_in[5];  const float* bk = (const float*)d_in[6];
  const float* Wv = (const float*)d_in[7];  const float* bv = (const float*)d_in[8];
  const float* Wo = (const float*)d_in[9];  const float* bo = (const float*)d_in[10];
  const long long* sidx = (const long long*)d_in[11];

  float* out  = (float*)d_out;                       // (B, L, D)
  float* attn = out + (size_t)B_ * L_ * D_;          // (B, H, U, L)

  const size_t NTOK = (size_t)B_ * L_ * D_;          // 16,777,216 floats
  float* ws  = (float*)d_ws;
  float* Qb  = ws;
  float* Kb  = Qb + NTOK;
  float* Vb  = Kb + NTOK;
  float* Pb  = Vb + NTOK;
  float* Mb  = Pb + NTOK;                            // B*H*L
  float* Vme = Mb + (size_t)BH_ * L_;                // B*H*DH
  float* Ctx = Vme + (size_t)BH_ * DH_;              // B*H*U*DH
  int*   Mtp = (int*)(Ctx + (size_t)BH_ * U_ * DH_); // B*H*U ints

  dim3 gGemm(GN / BN, GM / BM);                      // (16, 128)

  // Q/K/V projections (bf16 WMMA, fp32 accumulate, async-LDS staging)
  gemm_bf16_wmma<<<gGemm, 256, 0, stream>>>(queries, Wq, bq, Qb);
  gemm_bf16_wmma<<<gGemm, 256, 0, stream>>>(keys,    Wk, bk, Kb);
  gemm_bf16_wmma<<<gGemm, 256, 0, stream>>>(values,  Wv, bv, Vb);

  // ProbSparse selection
  sparsity_measure<<<dim3(L_ / 256, BH_), 256, 0, stream>>>(Qb, Kb, sidx, Mb);
  topk_u<<<BH_, 256, 0, stream>>>(Mb, Mtp);

  // Attention on selected queries (raw scores -> softmax in attn output region)
  scores_kernel<<<dim3(L_ / 256, BH_), 256, 0, stream>>>(Qb, Kb, Mtp, attn);
  softmax_kernel<<<BH_ * U_, 256, 0, stream>>>(attn);
  context_kernel<<<dim3(U_, BH_), 256, 0, stream>>>(attn, Vb, Ctx);

  // Assemble pre-projection output: V-mean everywhere, contexts scattered in
  vmean_kernel<<<BH_, 256, 0, stream>>>(Vb, Vme);
  fill_mean<<<(int)(NTOK / 256), 256, 0, stream>>>(Vme, Pb);
  scatter_ctx<<<(BH_ * U_ * DH_ + 255) / 256, 256, 0, stream>>>(Ctx, Mtp, Pb);

  // Output projection straight into d_out
  gemm_bf16_wmma<<<gGemm, 256, 0, stream>>>(Pb, Wo, bo, out);
}
